// Codebook_51342039056630
// MI455X (gfx1250) — compile-verified, matrix operand order checked by reading, not run on hardware
//
#include <hip/hip_runtime.h>

// VQ-VAE codebook quantization for MI455X (gfx1250, wave32, WMMA).
// z: (65536, 64) f32, embeddings: (1024, 64) f32.
// out = [quantized (65536*64 f32) | indices (65536, as f32)].

typedef __attribute__((ext_vector_type(16))) _Float16 v16h;
typedef __attribute__((ext_vector_type(8)))  _Float16 v8h;
typedef __attribute__((ext_vector_type(8)))  float    v8f;

#define KCODES      1024
#define DDIM        64
#define NROWS       65536          // 64*32*32
#define THREADS     256            // 8 waves
#define ROWS_PER_WG 128            // 8 waves * 16 rows
#define NGRID       (NROWS / ROWS_PER_WG)   // 512

// Pack 16 halves (two 8-K runs, 16 apart) of one A/B fragment row from four float4s.
__device__ inline v16h pack16f(float4 q0, float4 q1, float4 q2, float4 q3) {
  v16h a;
  a[0]  = (_Float16)q0.x; a[1]  = (_Float16)q0.y; a[2]  = (_Float16)q0.z; a[3]  = (_Float16)q0.w;
  a[4]  = (_Float16)q1.x; a[5]  = (_Float16)q1.y; a[6]  = (_Float16)q1.z; a[7]  = (_Float16)q1.w;
  a[8]  = (_Float16)q2.x; a[9]  = (_Float16)q2.y; a[10] = (_Float16)q2.z; a[11] = (_Float16)q2.w;
  a[12] = (_Float16)q3.x; a[13] = (_Float16)q3.y; a[14] = (_Float16)q3.z; a[15] = (_Float16)q3.w;
  return a;
}

__device__ inline v16h pack16h(v8h lo, v8h hi) {
  v16h b;
#pragma unroll
  for (int j = 0; j < 8; ++j) { b[j] = lo[j]; b[8 + j] = hi[j]; }
  return b;
}

// Fetch the eight 16B LDS vectors making up the B fragments of one
// 2-tile (32 codes) chunk pair: codes [cpair*16, cpair*16+32).
__device__ inline void loadB(const _Float16* __restrict__ semb,
                             int cpair, int n, int kb, v8h buf[8]) {
  const v8h* eb0 = (const v8h*)(semb + (size_t)(cpair * 16 + n) * DDIM);
  const v8h* eb1 = (const v8h*)(semb + (size_t)(cpair * 16 + 16 + n) * DDIM);
  buf[0] = eb0[kb >> 3];        buf[1] = eb0[(16 + kb) >> 3];
  buf[2] = eb0[(32 + kb) >> 3]; buf[3] = eb0[(48 + kb) >> 3];
  buf[4] = eb1[kb >> 3];        buf[5] = eb1[(16 + kb) >> 3];
  buf[6] = eb1[(32 + kb) >> 3]; buf[7] = eb1[(48 + kb) >> 3];
}

extern "C" __global__ __launch_bounds__(THREADS)
void vq_wmma_kernel(const float* __restrict__ z,
                    const float* __restrict__ emb,
                    float* __restrict__ out_q,
                    float* __restrict__ out_idx) {
  extern __shared__ char smem_raw[];
  _Float16* semb  = (_Float16*)smem_raw;                                   // 1024*64 f16 = 128 KB
  float*    senrm = (float*)(smem_raw + (size_t)KCODES * DDIM * 2);        // 1024 f32  =   4 KB

  // ---- Phase 0: stage f16 embedding table + squared norms into LDS ----
  for (int k = threadIdx.x; k < KCODES; k += THREADS) {
    const float4* e4 = (const float4*)(emb + (size_t)k * DDIM);
    float nrm = 0.0f;
    v8h hbuf[8];
#pragma unroll
    for (int j = 0; j < 16; ++j) {
      float4 v = e4[j];
      nrm += v.x * v.x + v.y * v.y + v.z * v.z + v.w * v.w;
      int o = (j & 1) * 4;
      hbuf[j >> 1][o + 0] = (_Float16)v.x;
      hbuf[j >> 1][o + 1] = (_Float16)v.y;
      hbuf[j >> 1][o + 2] = (_Float16)v.z;
      hbuf[j >> 1][o + 3] = (_Float16)v.w;
    }
    v8h* dst = (v8h*)(semb + (size_t)k * DDIM);
#pragma unroll
    for (int t = 0; t < 8; ++t) dst[t] = hbuf[t];
    senrm[k] = nrm;
  }
  __syncthreads();

  // ---- Phase 1: per-wave 16-row tile, scan all 1024 codes ----
  const int lane = threadIdx.x & 31;
  const int wave = threadIdx.x >> 5;
  const int n    = lane & 15;        // column-within-tile (code), also A-row M
  const int half = lane >> 4;        // 0 or 1
  const int kb   = half * 8;         // K sub-run base per ISA 16-bit A/B layout
  const int rowbase = blockIdx.x * ROWS_PER_WG + wave * 16;

  // A fragments: rows rowbase..rowbase+15 of z, K = 0..31 and 32..63.
  const float4* zr = (const float4*)(z + (size_t)(rowbase + n) * DDIM);
  v16h a0, a1;
  {
    a0 = pack16f(zr[kb >> 2], zr[(kb >> 2) + 1],
                 zr[(16 + kb) >> 2], zr[((16 + kb) >> 2) + 1]);
    a1 = pack16f(zr[(32 + kb) >> 2], zr[((32 + kb) >> 2) + 1],
                 zr[(48 + kb) >> 2], zr[((48 + kb) >> 2) + 1]);
  }

  float minv[8];
  int   mini[8];
#pragma unroll
  for (int r = 0; r < 8; ++r) { minv[r] = 3.4e38f; mini[r] = 0; }

  // Software-pipelined scan with interleaved accumulator chains:
  //   wmma acc0_k0, wmma acc1_k0, wmma acc0_k1, wmma acc1_k1
  // Each C-chained pair is separated by the other accumulator's (multi-cycle
  // XDL) WMMA, satisfying the WMMA->WMMA RAW spacing without v_nops. The
  // next chunk-pair's B fragments are prefetched from LDS between the WMMAs
  // and the accumulator consumers, covering the WMMA->VALU hazard and hiding
  // LDS latency under the ~40-op min-update block.
  v8h buf[8];
  loadB(semb, 0, n, kb, buf);

  for (int c = 0; c < 64; c += 2) {
    v16h b00 = pack16h(buf[0], buf[1]);
    v16h b01 = pack16h(buf[2], buf[3]);
    v16h b10 = pack16h(buf[4], buf[5]);
    v16h b11 = pack16h(buf[6], buf[7]);

    v8f acc0 = {};
    v8f acc1 = {};
    acc0 = __builtin_amdgcn_wmma_f32_16x16x32_f16(false, a0, false, b00,
                                                  (short)0, acc0, false, false);
    acc1 = __builtin_amdgcn_wmma_f32_16x16x32_f16(false, a0, false, b10,
                                                  (short)0, acc1, false, false);
    acc0 = __builtin_amdgcn_wmma_f32_16x16x32_f16(false, a1, false, b01,
                                                  (short)0, acc0, false, false);
    acc1 = __builtin_amdgcn_wmma_f32_16x16x32_f16(false, a1, false, b11,
                                                  (short)0, acc1, false, false);

    // Prefetch next chunk pair (wraps on the last iteration; harmless reload).
    loadB(semb, (c + 2) & 63, n, kb, buf);

    const int  code0 = c * 16 + n;
    const int  code1 = code0 + 16;
    const float en0 = senrm[code0];
    const float en1 = senrm[code1];
#pragma unroll
    for (int r = 0; r < 8; ++r) {
      float s0 = en0 - 2.0f * acc0[r];    // ||e||^2 - 2 z.e  (||z||^2 dropped)
      if (s0 < minv[r]) { minv[r] = s0; mini[r] = code0; }  // '<': first-code ties
    }
#pragma unroll
    for (int r = 0; r < 8; ++r) {
      float s1 = en1 - 2.0f * acc1[r];
      if (s1 < minv[r]) { minv[r] = s1; mini[r] = code1; }
    }
  }

  // ---- Reduce across the 16 lanes of each half; gather + write outputs ----
#pragma unroll
  for (int r = 0; r < 8; ++r) {
    float v = minv[r];
    int   i = mini[r];
#pragma unroll
    for (int off = 1; off < 16; off <<= 1) {
      float ov = __shfl_xor(v, off, 32);
      int   oi = __shfl_xor(i, off, 32);
      if (ov < v || (ov == v && oi < i)) { v = ov; i = oi; }  // argmin, first-occurrence ties
    }
    const int row = rowbase + r + half * 8;      // C layout: VGPR r <-> M = r + 8*half
    if (n == 0) out_idx[row] = (float)i;
    // 16 lanes of the half copy the winning f32 embedding row (coalesced float4s)
    const float4* src = (const float4*)(emb + (size_t)i * DDIM);
    float4*       dst = (float4*)(out_q + (size_t)row * DDIM);
    dst[n] = src[n];
  }
}

extern "C" void kernel_launch(void* const* d_in, const int* in_sizes, int n_in,
                              void* d_out, int out_size, void* d_ws, size_t ws_size,
                              hipStream_t stream) {
  (void)in_sizes; (void)n_in; (void)out_size; (void)d_ws; (void)ws_size;
  const float* z   = (const float*)d_in[0];
  const float* emb = (const float*)d_in[1];
  float* out_q   = (float*)d_out;
  float* out_idx = (float*)d_out + (size_t)NROWS * DDIM;

  const size_t lds_bytes = (size_t)KCODES * DDIM * 2 + (size_t)KCODES * 4;  // 132 KB

  // Opt into >64KB dynamic LDS (host-side attribute; deterministic, capture-safe).
  hipFuncSetAttribute((const void*)vq_wmma_kernel,
                      hipFuncAttributeMaxDynamicSharedMemorySize, (int)lds_bytes);

  hipLaunchKernelGGL(vq_wmma_kernel, dim3(NGRID), dim3(THREADS), lds_bytes, stream,
                     z, emb, out_q, out_idx);
}
